// FusionMoE_85495618994888
// MI455X (gfx1250) — matrix operand order repaired and checked by reading, not compiled, not run on hardware
//
#include <hip/hip_runtime.h>
#include <math.h>

#define B_TOK 131072
#define DMODEL 256

typedef __bf16 bf16_t;
typedef __attribute__((ext_vector_type(16))) __bf16 v16bf;
typedef __attribute__((ext_vector_type(8)))  __bf16 v8bf;
typedef __attribute__((ext_vector_type(8)))  float  v8f;

#define DINL static __device__ __forceinline__

// ---------------- WMMA helpers ----------------

DINL v8f wmma_bf(v16bf a, v16bf b, v8f c) {
  // (neg_a, A, neg_b, B, c_mod, C, reuse_a, reuse_b)
  return __builtin_amdgcn_wmma_f32_16x16x32_bf16(false, a, false, b, (short)0, c, false, false);
}

DINL void cvt8(float4 f0, float4 f1, bf16_t* d) {
  d[0] = (bf16_t)f0.x; d[1] = (bf16_t)f0.y; d[2] = (bf16_t)f0.z; d[3] = (bf16_t)f0.w;
  d[4] = (bf16_t)f1.x; d[5] = (bf16_t)f1.y; d[6] = (bf16_t)f1.z; d[7] = (bf16_t)f1.w;
}

// A-fragment (16x32 bf16) for x = [r0 | r1] (K up to 512). Lane holds row m=lane&15,
// k runs [c0,c0+8) and [c0+16,c0+24), c0 = k0 + (lane>=16 ? 8 : 0). Runs never cross 256.
DINL v16bf afrag_x(const float* r0, const float* r1, int k0, int lane) {
  bf16_t tmp[16];
  int c0 = k0 + ((lane & 16) ? 8 : 0);
  {
    const float* p = (c0 < 256) ? (r0 + c0) : (r1 + (c0 - 256));
    cvt8(*(const float4*)p, *(const float4*)(p + 4), tmp);
  }
  {
    int c1 = c0 + 16;
    const float* p = (c1 < 256) ? (r0 + c1) : (r1 + (c1 - 256));
    cvt8(*(const float4*)p, *(const float4*)(p + 4), tmp + 8);
  }
  v16bf a;
#pragma unroll
  for (int i = 0; i < 16; ++i) a[i] = tmp[i];
  return a;
}

// A-fragment for elementwise product z_gat*z_gin (K=256)
DINL v16bf afrag_prod(const float* r0, const float* r1, int k0, int lane) {
  bf16_t tmp[16];
  int c0 = k0 + ((lane & 16) ? 8 : 0);
#pragma unroll
  for (int half = 0; half < 2; ++half) {
    int c = c0 + half * 16;
    float4 g0 = *(const float4*)(r0 + c);
    float4 g1 = *(const float4*)(r0 + c + 4);
    float4 i0 = *(const float4*)(r1 + c);
    float4 i1 = *(const float4*)(r1 + c + 4);
    bf16_t* d = tmp + half * 8;
    d[0] = (bf16_t)(g0.x * i0.x); d[1] = (bf16_t)(g0.y * i0.y);
    d[2] = (bf16_t)(g0.z * i0.z); d[3] = (bf16_t)(g0.w * i0.w);
    d[4] = (bf16_t)(g1.x * i1.x); d[5] = (bf16_t)(g1.y * i1.y);
    d[6] = (bf16_t)(g1.z * i1.z); d[7] = (bf16_t)(g1.w * i1.w);
  }
  v16bf a;
#pragma unroll
  for (int i = 0; i < 16; ++i) a[i] = tmp[i];
  return a;
}

// A-fragment from an LDS bf16 buffer (16 x 256)
DINL v16bf afrag_lds(const bf16_t* h, int k0, int lane) {
  const bf16_t* row = h + (lane & 15) * 256;
  int c0 = k0 + ((lane & 16) ? 8 : 0);
  v8bf r0 = *(const v8bf*)(row + c0);
  v8bf r1 = *(const v8bf*)(row + c0 + 16);
  v16bf a;
#pragma unroll
  for (int i = 0; i < 8; ++i) { a[i] = r0[i]; a[8 + i] = r1[i]; }
  return a;
}

// B-fragment from packed weights: one contiguous 32B per lane
DINL v16bf bfrag(const bf16_t* pw, int numKt, int nt, int kt, int lane) {
  return *(const v16bf*)(pw + (((size_t)(nt * numKt + kt)) * 32 + lane) * 16);
}

DINL float sigmoidf_(float x) { return 1.0f / (1.0f + __expf(-x)); }

// ---------------- meta kernels ----------------

__global__ void zero_meta_kernel(int* cnt, unsigned long long* psum, float* avg) {
  int t = threadIdx.x;
  if (t < 4) { cnt[t] = 0; psum[t] = 0ull; avg[t] = 0.f; }
}

__global__ void __launch_bounds__(256) gate_kernel(
    const float* __restrict__ zg, const float* __restrict__ zi,
    const float* __restrict__ gW, const float* __restrict__ gb,
    int* cnt, unsigned long long* psum, int* lists) {
  __shared__ float sW[2048];
  __shared__ unsigned long long sps[4];
  int t = threadIdx.x;
  for (int i = t; i < 2048; i += 256) sW[i] = gW[i];
  if (t < 4) sps[t] = 0ull;
  __syncthreads();
  int tok = blockIdx.x * 256 + t;
  const float* a = zg + (size_t)tok * 256;
  const float* b = zi + (size_t)tok * 256;
  float l0 = gb[0], l1 = gb[1], l2 = gb[2], l3 = gb[3];
  for (int k = 0; k < 256; k += 4) {
    float4 xa = *(const float4*)(a + k);
    float4 xb = *(const float4*)(b + k);
    const float* xaf = (const float*)&xa;
    const float* xbf = (const float*)&xb;
#pragma unroll
    for (int u = 0; u < 4; ++u) {
      const float* w0 = sW + (k + u) * 4;
      const float* w1 = sW + (256 + k + u) * 4;
      float va = xaf[u], vb = xbf[u];
      l0 += va * w0[0] + vb * w1[0];
      l1 += va * w0[1] + vb * w1[1];
      l2 += va * w0[2] + vb * w1[2];
      l3 += va * w0[3] + vb * w1[3];
    }
  }
  int idx = 0; float best = l0;
  if (l1 > best) { best = l1; idx = 1; }
  if (l2 > best) { best = l2; idx = 2; }
  if (l3 > best) { best = l3; idx = 3; }
  float e0 = __expf(l0 - best), e1 = __expf(l1 - best);
  float e2 = __expf(l2 - best), e3 = __expf(l3 - best);
  float s = e0 + e1 + e2 + e3;
  float pe = (idx == 0 ? e0 : idx == 1 ? e1 : idx == 2 ? e2 : e3) / s;
  int pos = atomicAdd(&cnt[idx], 1);
  lists[(size_t)idx * B_TOK + pos] = tok;
  atomicAdd(&sps[idx], (unsigned long long)((double)pe * 4294967296.0));
  __syncthreads();
  if (t < 4) atomicAdd(&psum[t], sps[t]);
}

__global__ void finalize_kernel(const int* cnt, const unsigned long long* psum,
                                float* avg, float* aux_out) {
  if (threadIdx.x == 0) {
    float aux = 0.f;
    for (int e = 0; e < 4; ++e) {
      int c = cnt[e];
      double s = (double)psum[e] * (1.0 / 4294967296.0);
      avg[e] = (c > 0) ? (float)(s / (double)c) : 0.f;
      float u = (float)c / (float)B_TOK;
      aux += u * u;
    }
    *aux_out = aux * 4.0f;
  }
}

// fp32 row-major (K x N) -> bf16 WMMA-B fragment layout
__global__ void pack_kernel(const float* __restrict__ W, bf16_t* __restrict__ dst,
                            int K, int N) {
  int i = blockIdx.x * 256 + threadIdx.x;
  if (i >= K * N) return;
  int numKt = K >> 5;
  int j = i & 15;
  int lane = (i >> 4) & 31;
  int kt = (i >> 9) % numKt;
  int nt = (i >> 9) / numKt;
  int k = kt * 32 + ((lane & 16) ? 8 : 0) + (j & 7) + ((j & 8) ? 16 : 0);
  int n = nt * 16 + (lane & 15);
  dst[i] = (bf16_t)W[(size_t)k * N + n];
}

// ---------------- expert tile prologue ----------------
#define TILE_PROLOGUE(cntp, listp)                                              \
  __shared__ int srows[16]; __shared__ int sc;                                  \
  int t = threadIdx.x;                                                          \
  if (t == 0) sc = *(cntp);                                                     \
  __syncthreads();                                                              \
  int c_ = sc; int base_ = blockIdx.x * 16;                                     \
  if (base_ >= c_) return;                                                      \
  int valid = c_ - base_; if (valid > 16) valid = 16;                           \
  if (t < 16) { int j = base_ + t; if (j >= c_) j = c_ - 1; srows[t] = (listp)[j]; } \
  __syncthreads();                                                              \
  int lane = t & 31, wv = t >> 5;                                               \
  int nt0 = wv, nt1 = wv + 8;                                                   \
  int mbase = (lane & 16) ? 8 : 0;                                              \
  int nl0 = nt0 * 16 + (lane & 15), nl1 = nt1 * 16 + (lane & 15);

// ---------------- expert: concat ----------------
__global__ void __launch_bounds__(256) expert_concat_kernel(
    const float* __restrict__ zg, const float* __restrict__ zi,
    const bf16_t* __restrict__ pW1, const float* __restrict__ b1,
    const bf16_t* __restrict__ pW2, const float* __restrict__ b2,
    const int* cnt, const int* list, const float* avg_e, float* __restrict__ out) {
  __shared__ bf16_t sh[16 * 256];
  TILE_PROLOGUE(cnt, list)
  int row = srows[lane & 15];
  const float* rg = zg + (size_t)row * 256;
  const float* ri = zi + (size_t)row * 256;
  v8f acc0 = {}, acc1 = {};
  for (int kt = 0; kt < 16; ++kt) {
    v16bf a = afrag_x(rg, ri, kt * 32, lane);
    acc0 = wmma_bf(a, bfrag(pW1, 16, nt0, kt, lane), acc0);
    acc1 = wmma_bf(a, bfrag(pW1, 16, nt1, kt, lane), acc1);
  }
  float bi0 = b1[nl0], bi1 = b1[nl1];
#pragma unroll
  for (int r = 0; r < 8; ++r) {
    float v0 = acc0[r] + bi0; v0 = v0 > 0.f ? v0 : 0.f;
    float v1 = acc1[r] + bi1; v1 = v1 > 0.f ? v1 : 0.f;
    sh[(mbase + r) * 256 + nl0] = (bf16_t)v0;
    sh[(mbase + r) * 256 + nl1] = (bf16_t)v1;
  }
  __syncthreads();
  v8f d0 = {}, d1 = {};
  for (int kt = 0; kt < 8; ++kt) {
    v16bf a = afrag_lds(sh, kt * 32, lane);
    d0 = wmma_bf(a, bfrag(pW2, 8, nt0, kt, lane), d0);
    d1 = wmma_bf(a, bfrag(pW2, 8, nt1, kt, lane), d1);
  }
  float sc2 = *avg_e;
  float bo0 = b2[nl0], bo1 = b2[nl1];
#pragma unroll
  for (int r = 0; r < 8; ++r) {
    int m = mbase + r;
    if (m < valid) {
      size_t o = (size_t)srows[m] * 256;
      out[o + nl0] = sc2 * (d0[r] + bo0);
      out[o + nl1] = sc2 * (d1[r] + bo1);
    }
  }
}

// ---------------- expert: mul ----------------
__global__ void __launch_bounds__(256) expert_mul_kernel(
    const float* __restrict__ zg, const float* __restrict__ zi,
    const bf16_t* __restrict__ pW1, const float* __restrict__ b1,
    const bf16_t* __restrict__ pW2, const float* __restrict__ b2,
    const int* cnt, const int* list, const float* avg_e, float* __restrict__ out) {
  __shared__ bf16_t sh[16 * 256];
  TILE_PROLOGUE(cnt, list)
  int row = srows[lane & 15];
  const float* rg = zg + (size_t)row * 256;
  const float* ri = zi + (size_t)row * 256;
  v8f acc0 = {}, acc1 = {};
  for (int kt = 0; kt < 8; ++kt) {
    v16bf a = afrag_prod(rg, ri, kt * 32, lane);
    acc0 = wmma_bf(a, bfrag(pW1, 8, nt0, kt, lane), acc0);
    acc1 = wmma_bf(a, bfrag(pW1, 8, nt1, kt, lane), acc1);
  }
  float bi0 = b1[nl0], bi1 = b1[nl1];
#pragma unroll
  for (int r = 0; r < 8; ++r) {
    float v0 = acc0[r] + bi0; v0 = v0 > 0.f ? v0 : 0.f;
    float v1 = acc1[r] + bi1; v1 = v1 > 0.f ? v1 : 0.f;
    sh[(mbase + r) * 256 + nl0] = (bf16_t)v0;
    sh[(mbase + r) * 256 + nl1] = (bf16_t)v1;
  }
  __syncthreads();
  v8f d0 = {}, d1 = {};
  for (int kt = 0; kt < 8; ++kt) {
    v16bf a = afrag_lds(sh, kt * 32, lane);
    d0 = wmma_bf(a, bfrag(pW2, 8, nt0, kt, lane), d0);
    d1 = wmma_bf(a, bfrag(pW2, 8, nt1, kt, lane), d1);
  }
  float sc2 = *avg_e;
  float bo0 = b2[nl0], bo1 = b2[nl1];
#pragma unroll
  for (int r = 0; r < 8; ++r) {
    int m = mbase + r;
    if (m < valid) {
      size_t o = (size_t)srows[m] * 256;
      out[o + nl0] = sc2 * (d0[r] + bo0);
      out[o + nl1] = sc2 * (d1[r] + bo1);
    }
  }
}

// ---------------- expert: wsum ----------------
__global__ void __launch_bounds__(256) expert_wsum_kernel(
    const float* __restrict__ zg, const float* __restrict__ zi,
    const bf16_t* __restrict__ pWa, const float* __restrict__ ba,
    const bf16_t* __restrict__ pWo, const float* __restrict__ bo,
    const int* cnt, const int* list, const float* avg_e, float* __restrict__ out) {
  __shared__ bf16_t sh[16 * 256];
  TILE_PROLOGUE(cnt, list)
  int row = srows[lane & 15];
  const float* rg = zg + (size_t)row * 256;
  const float* ri = zi + (size_t)row * 256;
  v8f acc0 = {}, acc1 = {};
  for (int kt = 0; kt < 16; ++kt) {
    v16bf a = afrag_x(rg, ri, kt * 32, lane);
    acc0 = wmma_bf(a, bfrag(pWa, 16, nt0, kt, lane), acc0);
    acc1 = wmma_bf(a, bfrag(pWa, 16, nt1, kt, lane), acc1);
  }
  float bi0 = ba[nl0], bi1 = ba[nl1];
#pragma unroll
  for (int r = 0; r < 8; ++r) {
    int m = mbase + r;
    size_t ro = (size_t)srows[m] * 256;
    float a0 = sigmoidf_(acc0[r] + bi0);
    float a1 = sigmoidf_(acc1[r] + bi1);
    float g0 = zg[ro + nl0], i0 = zi[ro + nl0];
    float g1 = zg[ro + nl1], i1 = zi[ro + nl1];
    sh[m * 256 + nl0] = (bf16_t)(a0 * g0 + (1.f - a0) * i0);
    sh[m * 256 + nl1] = (bf16_t)(a1 * g1 + (1.f - a1) * i1);
  }
  __syncthreads();
  v8f d0 = {}, d1 = {};
  for (int kt = 0; kt < 8; ++kt) {
    v16bf a = afrag_lds(sh, kt * 32, lane);
    d0 = wmma_bf(a, bfrag(pWo, 8, nt0, kt, lane), d0);
    d1 = wmma_bf(a, bfrag(pWo, 8, nt1, kt, lane), d1);
  }
  float sc2 = *avg_e;
  float bo0 = bo[nl0], bo1 = bo[nl1];
#pragma unroll
  for (int r = 0; r < 8; ++r) {
    int m = mbase + r;
    if (m < valid) {
      size_t o = (size_t)srows[m] * 256;
      out[o + nl0] = sc2 * (d0[r] + bo0);
      out[o + nl1] = sc2 * (d1[r] + bo1);
    }
  }
}

// ---------------- expert: attention ----------------
// K=256 projection GEMM into an LDS bf16 buffer (uses prologue locals)
#define PROJ256(dst, srcrow, pW, bias) do {                                     \
    v8f a0_ = {}, a1_ = {};                                                     \
    for (int kt = 0; kt < 8; ++kt) {                                            \
      v16bf af_ = afrag_x((srcrow), (srcrow), kt * 32, lane);                   \
      a0_ = wmma_bf(af_, bfrag((pW), 8, nt0, kt, lane), a0_);                   \
      a1_ = wmma_bf(af_, bfrag((pW), 8, nt1, kt, lane), a1_);                   \
    }                                                                           \
    float bi0_ = (bias)[nl0], bi1_ = (bias)[nl1];                               \
    _Pragma("unroll")                                                           \
    for (int r = 0; r < 8; ++r) {                                               \
      (dst)[(mbase + r) * 256 + nl0] = (bf16_t)(a0_[r] + bi0_);                 \
      (dst)[(mbase + r) * 256 + nl1] = (bf16_t)(a1_[r] + bi1_);                 \
    }                                                                           \
  } while (0)

__global__ void __launch_bounds__(256) expert_attn_kernel(
    const float* __restrict__ zg, const float* __restrict__ zi,
    const bf16_t* __restrict__ pWq, const float* __restrict__ bq,
    const bf16_t* __restrict__ pWk, const float* __restrict__ bk,
    const bf16_t* __restrict__ pWv, const float* __restrict__ bv,
    const bf16_t* __restrict__ pWo, const float* __restrict__ bo,
    const bf16_t* __restrict__ pWfc, const float* __restrict__ bfc,
    const int* cnt, const int* list, const float* avg_e, float* __restrict__ out) {
  __shared__ bf16_t sq0[4096], sq1[4096], sk0[4096], sk1[4096], sv0[4096], sv1[4096];
  __shared__ float sw0[64], sw1[64];
  TILE_PROLOGUE(cnt, list)
  int row = srows[lane & 15];
  const float* rg = zg + (size_t)row * 256;
  const float* ri = zi + (size_t)row * 256;
  // Q/K/V projections for both sequence positions
  PROJ256(sq0, rg, pWq, bq);
  PROJ256(sq1, ri, pWq, bq);
  PROJ256(sk0, rg, pWk, bk);
  PROJ256(sk1, ri, pWk, bk);
  PROJ256(sv0, rg, pWv, bv);
  PROJ256(sv1, ri, pWv, bv);
  __syncthreads();
  // scores + softmax over seqlen=2, folded with mean over positions:
  // obar = w0*V0 + w1*V1 with w0=(a00+a10)/2, w1=(a01+a11)/2
  if (t < 64) {
    int m = t & 15, h = t >> 4;
    const bf16_t* q0 = sq0 + m * 256 + h * 64;
    const bf16_t* q1 = sq1 + m * 256 + h * 64;
    const bf16_t* k0 = sk0 + m * 256 + h * 64;
    const bf16_t* k1 = sk1 + m * 256 + h * 64;
    float s00 = 0.f, s01 = 0.f, s10 = 0.f, s11 = 0.f;
    for (int i = 0; i < 64; ++i) {
      float a0 = (float)q0[i], a1 = (float)q1[i];
      float b0 = (float)k0[i], b1 = (float)k1[i];
      s00 += a0 * b0; s01 += a0 * b1; s10 += a1 * b0; s11 += a1 * b1;
    }
    s00 *= 0.125f; s01 *= 0.125f; s10 *= 0.125f; s11 *= 0.125f;
    float mx0 = fmaxf(s00, s01);
    float e00 = __expf(s00 - mx0), e01 = __expf(s01 - mx0);
    float r0 = 1.f / (e00 + e01);
    float mx1 = fmaxf(s10, s11);
    float e10 = __expf(s10 - mx1), e11 = __expf(s11 - mx1);
    float r1 = 1.f / (e10 + e11);
    sw0[t] = 0.5f * (e00 * r0 + e10 * r1);
    sw1[t] = 0.5f * (e01 * r0 + e11 * r1);
  }
  __syncthreads();
  bf16_t* sob = sq0;  // reuse: scores consumed
  bf16_t* sh2 = sq1;
  for (int it = 0; it < 16; ++it) {
    int e = it * 256 + t;
    int m = e >> 8, n = e & 255, h = n >> 6;
    float w0 = sw0[h * 16 + m], w1 = sw1[h * 16 + m];
    sob[e] = (bf16_t)(w0 * (float)sv0[e] + w1 * (float)sv1[e]);
  }
  __syncthreads();
  {
    v8f a0 = {}, a1 = {};
    for (int kt = 0; kt < 8; ++kt) {
      v16bf af = afrag_lds(sob, kt * 32, lane);
      a0 = wmma_bf(af, bfrag(pWo, 8, nt0, kt, lane), a0);
      a1 = wmma_bf(af, bfrag(pWo, 8, nt1, kt, lane), a1);
    }
    float bi0 = bo[nl0], bi1 = bo[nl1];
#pragma unroll
    for (int r = 0; r < 8; ++r) {
      sh2[(mbase + r) * 256 + nl0] = (bf16_t)(a0[r] + bi0);
      sh2[(mbase + r) * 256 + nl1] = (bf16_t)(a1[r] + bi1);
    }
  }
  __syncthreads();
  {
    v8f a0 = {}, a1 = {};
    for (int kt = 0; kt < 8; ++kt) {
      v16bf af = afrag_lds(sh2, kt * 32, lane);
      a0 = wmma_bf(af, bfrag(pWfc, 8, nt0, kt, lane), a0);
      a1 = wmma_bf(af, bfrag(pWfc, 8, nt1, kt, lane), a1);
    }
    float sc2 = *avg_e;
    float bi0 = bfc[nl0], bi1 = bfc[nl1];
#pragma unroll
    for (int r = 0; r < 8; ++r) {
      int m = mbase + r;
      if (m < valid) {
        size_t o = (size_t)srows[m] * 256;
        out[o + nl0] = sc2 * (a0[r] + bi0);
        out[o + nl1] = sc2 * (a1[r] + bi1);
      }
    }
  }
}

// ---------------- host launch ----------------
extern "C" void kernel_launch(void* const* d_in, const int* in_sizes, int n_in,
                              void* d_out, int out_size, void* d_ws, size_t ws_size,
                              hipStream_t stream) {
  const float* zg     = (const float*)d_in[0];
  const float* zi     = (const float*)d_in[1];
  const float* gate_W = (const float*)d_in[2];
  const float* gate_b = (const float*)d_in[3];
  const float* c_W1 = (const float*)d_in[4];  const float* c_b1 = (const float*)d_in[5];
  const float* c_W2 = (const float*)d_in[6];  const float* c_b2 = (const float*)d_in[7];
  const float* m_W1 = (const float*)d_in[8];  const float* m_b1 = (const float*)d_in[9];
  const float* m_W2 = (const float*)d_in[10]; const float* m_b2 = (const float*)d_in[11];
  const float* a_Wq = (const float*)d_in[12]; const float* a_bq = (const float*)d_in[13];
  const float* a_Wk = (const float*)d_in[14]; const float* a_bk = (const float*)d_in[15];
  const float* a_Wv = (const float*)d_in[16]; const float* a_bv = (const float*)d_in[17];
  const float* a_Wo = (const float*)d_in[18]; const float* a_bo = (const float*)d_in[19];
  const float* a_Wfc = (const float*)d_in[20]; const float* a_bfc = (const float*)d_in[21];
  const float* w_Wa = (const float*)d_in[22]; const float* w_ba = (const float*)d_in[23];
  const float* w_Wo = (const float*)d_in[24]; const float* w_bo = (const float*)d_in[25];

  float* out = (float*)d_out;
  float* aux = out + (size_t)B_TOK * DMODEL;

  char* ws = (char*)d_ws;
  int* cnt                  = (int*)(ws + 0);
  unsigned long long* psum  = (unsigned long long*)(ws + 64);
  float* avg                = (float*)(ws + 128);
  int* lists                = (int*)(ws + 256);
  char* pk = ws + 256 + (size_t)4 * B_TOK * 4;   // 256-byte aligned
  bf16_t* p_cW1 = (bf16_t*)pk; pk += (size_t)512 * 256 * 2;
  bf16_t* p_cW2 = (bf16_t*)pk; pk += (size_t)256 * 256 * 2;
  bf16_t* p_mW1 = (bf16_t*)pk; pk += (size_t)256 * 256 * 2;
  bf16_t* p_mW2 = (bf16_t*)pk; pk += (size_t)256 * 256 * 2;
  bf16_t* p_aWq = (bf16_t*)pk; pk += (size_t)256 * 256 * 2;
  bf16_t* p_aWk = (bf16_t*)pk; pk += (size_t)256 * 256 * 2;
  bf16_t* p_aWv = (bf16_t*)pk; pk += (size_t)256 * 256 * 2;
  bf16_t* p_aWo = (bf16_t*)pk; pk += (size_t)256 * 256 * 2;
  bf16_t* p_aWfc = (bf16_t*)pk; pk += (size_t)256 * 256 * 2;
  bf16_t* p_wWa = (bf16_t*)pk; pk += (size_t)512 * 256 * 2;
  bf16_t* p_wWo = (bf16_t*)pk; pk += (size_t)256 * 256 * 2;

  hipLaunchKernelGGL(zero_meta_kernel, dim3(1), dim3(64), 0, stream, cnt, psum, avg);
  hipLaunchKernelGGL(gate_kernel, dim3(B_TOK / 256), dim3(256), 0, stream,
                     zg, zi, gate_W, gate_b, cnt, psum, lists);
  hipLaunchKernelGGL(finalize_kernel, dim3(1), dim3(32), 0, stream, cnt, psum, avg, aux);

  auto packW = [&](const float* W, bf16_t* dst, int K, int N) {
    int total = K * N;
    hipLaunchKernelGGL(pack_kernel, dim3((total + 255) / 256), dim3(256), 0, stream,
                       W, dst, K, N);
  };
  packW(c_W1, p_cW1, 512, 256);
  packW(c_W2, p_cW2, 256, 256);
  packW(m_W1, p_mW1, 256, 256);
  packW(m_W2, p_mW2, 256, 256);
  packW(a_Wq, p_aWq, 256, 256);
  packW(a_Wk, p_aWk, 256, 256);
  packW(a_Wv, p_aWv, 256, 256);
  packW(a_Wo, p_aWo, 256, 256);
  packW(a_Wfc, p_aWfc, 256, 256);
  packW(w_Wa, p_wWa, 512, 256);
  packW(w_Wo, p_wWo, 256, 256);

  dim3 eg(B_TOK / 16), eb(256);
  hipLaunchKernelGGL(expert_concat_kernel, eg, eb, 0, stream,
                     zg, zi, p_cW1, c_b1, p_cW2, c_b2,
                     cnt + 0, lists + (size_t)0 * B_TOK, avg + 0, out);
  hipLaunchKernelGGL(expert_mul_kernel, eg, eb, 0, stream,
                     zg, zi, p_mW1, m_b1, p_mW2, m_b2,
                     cnt + 1, lists + (size_t)1 * B_TOK, avg + 1, out);
  hipLaunchKernelGGL(expert_attn_kernel, eg, eb, 0, stream,
                     zg, zi, p_aWq, a_bq, p_aWk, a_bk, p_aWv, a_bv,
                     p_aWo, a_bo, p_aWfc, a_bfc,
                     cnt + 2, lists + (size_t)2 * B_TOK, avg + 2, out);
  hipLaunchKernelGGL(expert_wsum_kernel, eg, eb, 0, stream,
                     zg, zi, p_wWa, w_ba, p_wWo, w_bo,
                     cnt + 3, lists + (size_t)3 * B_TOK, avg + 3, out);
  (void)in_sizes; (void)n_in; (void)out_size; (void)ws_size;
}
